// MambaModel_89558658056577
// MI455X (gfx1250) — matrix operand order repaired and checked by reading, not compiled, not run on hardware
//
#include <hip/hip_runtime.h>
#include <hip/hip_bf16.h>
#include <cstdint>

// ---------------- model dims ----------------
#define S_LEN  1024
#define DMODEL 768
#define DINNER 1536
#define DTRANK 48
#define DSTATE 16
#define DCONV  4
#define NVOCAB 32000
#define NLAYERS 4
#define PROJW  (DTRANK + 2 * DSTATE)   // 80
#define SCHUNK 32                      // scan chunk length
#define NCHUNK (S_LEN / SCHUNK)        // 32 chunks

typedef __attribute__((ext_vector_type(16))) _Float16 v16h;
typedef __attribute__((ext_vector_type(8)))  _Float16 v8h;
typedef __attribute__((ext_vector_type(8)))  float    v8f;
typedef __attribute__((ext_vector_type(4)))  unsigned int u32x4;
typedef __attribute__((ext_vector_type(4)))  int      i32x4;
typedef __attribute__((ext_vector_type(8)))  int      i32x8;

__device__ __forceinline__ float silu_f(float x)     { return x / (1.0f + __expf(-x)); }
__device__ __forceinline__ float softplus_f(float x) { return (x > 20.0f) ? x : log1pf(__expf(x)); }

// ---------------- WMMA GEMM:  C[M,N] = act(A[M,K] * B[N,K]^T + bias) (+ res) ----------------
// 128x128 block tile, BK=32, 8 waves (4 along M x 2 along N), each wave 32x64 (2x4 WMMA tiles).
// f16 operands (converted on stage-in), f32 accumulation via v_wmma_f32_16x16x32_f16.
// Double-buffered LDS with register pipelining: global->regs overlaps with WMMA compute.
constexpr int BM = 128, BN = 128, BK = 32;
constexpr int LDSS = BK + 16;            // 48 halves/row = 96 B -> 16B-aligned sub-chunks

__device__ __forceinline__ void stage_load16(const float* __restrict__ P, size_t rowoff,
                                             bool rowOk, int kc, int K, bool fullTile,
                                             float* __restrict__ out)
{
    if (rowOk && fullTile) {
        const float4* p4 = reinterpret_cast<const float4*>(P + rowoff + kc);
        #pragma unroll
        for (int v = 0; v < 4; ++v) {
            const float4 f = p4[v];
            out[4 * v + 0] = f.x; out[4 * v + 1] = f.y;
            out[4 * v + 2] = f.z; out[4 * v + 3] = f.w;
        }
    } else {
        #pragma unroll
        for (int i = 0; i < 16; ++i) {
            const int gk = kc + i;
            out[i] = (rowOk && gk < K) ? P[rowoff + gk] : 0.0f;
        }
    }
}

__device__ __forceinline__ void stage_store16(_Float16* __restrict__ dst,
                                              const float* __restrict__ src)
{
    v8h h0, h1;
    #pragma unroll
    for (int i = 0; i < 8; ++i) { h0[i] = (_Float16)src[i]; h1[i] = (_Float16)src[8 + i]; }
    *reinterpret_cast<v8h*>(dst)     = h0;
    *reinterpret_cast<v8h*>(dst + 8) = h1;
}

// ACT: 0 = none, 1 = softplus
template<int ACT, bool HAS_BIAS, bool HAS_RES>
__global__ __launch_bounds__(256)
void gemm_tn_wmma(const float* __restrict__ A, int lda,
                  const float* __restrict__ B, int ldb,
                  const float* __restrict__ bias,
                  const float* __restrict__ res, int ldres,
                  float* __restrict__ C, int ldc,
                  int M, int N, int K)
{
    __shared__ _Float16 sA[2][BM * LDSS];   // 2 x 12 KB
    __shared__ _Float16 sB[2][BN * LDSS];   // 2 x 12 KB

    const int tid    = threadIdx.x;
    const int lane   = tid & 31;
    const int lane16 = lane & 15;
    const int halfSel= lane >> 4;
    const int wave   = tid >> 5;
    const int waveM  = wave & 3;
    const int waveN  = wave >> 2;

    const int blockM = blockIdx.y * BM;
    const int blockN = blockIdx.x * BN;

    v8f acc[2][4];
    #pragma unroll
    for (int i = 0; i < 2; ++i)
        #pragma unroll
        for (int j = 0; j < 4; ++j)
            #pragma unroll
            for (int r = 0; r < 8; ++r) acc[i][j][r] = 0.0f;

    const int ar  = tid >> 1;            // 0..127
    const int ac0 = (tid & 1) << 4;      // 0 or 16
    const int gmA = blockM + ar;
    const int gnB = blockN + ar;
    const bool aRowOk = (gmA < M);
    const bool bRowOk = (gnB < N);
    const size_t aOff = (size_t)gmA * lda;
    const size_t bOff = (size_t)gnB * ldb;
    _Float16* const sAdst0 = &sA[0][ar * LDSS + ac0];
    _Float16* const sBdst0 = &sB[0][ar * LDSS + ac0];
    _Float16* const sAdst1 = &sA[1][ar * LDSS + ac0];
    _Float16* const sBdst1 = &sB[1][ar * LDSS + ac0];

    const int nt = (K + BK - 1) / BK;
    float aReg[16], bReg[16];

    stage_load16(A, aOff, aRowOk, ac0, K, BK <= K, aReg);
    stage_load16(B, bOff, bRowOk, ac0, K, BK <= K, bReg);
    stage_store16(sAdst0, aReg);
    stage_store16(sBdst0, bReg);
    __syncthreads();

    for (int kt = 0; kt < nt; ++kt) {
        const int cur = kt & 1;
        const bool haveNext = (kt + 1 < nt);

        if (haveNext) {
            const int k0n = (kt + 1) * BK;
            stage_load16(A, aOff, aRowOk, k0n + ac0, K, k0n + BK <= K, aReg);
            stage_load16(B, bOff, bRowOk, k0n + ac0, K, k0n + BK <= K, bReg);
            if (kt + 2 < nt) {
                __builtin_prefetch(A + aOff + (size_t)(kt + 2) * BK + ac0, 0, 1);
                __builtin_prefetch(B + bOff + (size_t)(kt + 2) * BK + ac0, 0, 1);
            }
        }

        {
            const _Float16* __restrict__ pa = sA[cur];
            const _Float16* __restrict__ pb = sB[cur];
            v16h afrag[2], bfrag[4];
            #pragma unroll
            for (int tm = 0; tm < 2; ++tm) {
                const _Float16* p = &pa[(waveM * 32 + tm * 16 + lane16) * LDSS + halfSel * 16];
                #pragma unroll
                for (int i = 0; i < 8; ++i) { afrag[tm][i] = p[i]; afrag[tm][8 + i] = p[8 + i]; }
            }
            #pragma unroll
            for (int tn = 0; tn < 4; ++tn) {
                const _Float16* p = &pb[(waveN * 64 + tn * 16 + lane16) * LDSS + halfSel * 16];
                #pragma unroll
                for (int i = 0; i < 16; ++i) bfrag[tn][i] = p[i];
            }
            #pragma unroll
            for (int tm = 0; tm < 2; ++tm)
                #pragma unroll
                for (int tn = 0; tn < 4; ++tn)
                    acc[tm][tn] = __builtin_amdgcn_wmma_f32_16x16x32_f16(
                        false, afrag[tm], false, bfrag[tn],
                        (short)0, acc[tm][tn], false, false);
        }

        if (haveNext) {
            stage_store16(cur ? sAdst0 : sAdst1, aReg);
            stage_store16(cur ? sBdst0 : sBdst1, bReg);
        }
        __syncthreads();
    }

    #pragma unroll
    for (int tm = 0; tm < 2; ++tm)
        #pragma unroll
        for (int tn = 0; tn < 4; ++tn)
            #pragma unroll
            for (int r = 0; r < 8; ++r) {
                const int row = blockM + waveM * 32 + tm * 16 + r + halfSel * 8;
                const int col = blockN + waveN * 64 + tn * 16 + lane16;
                if (row < M && col < N) {
                    float v = acc[tm][tn][r];
                    if (HAS_BIAS) v += bias[col];
                    if (ACT == 1) v = softplus_f(v);
                    if (HAS_RES)  v += res[(size_t)row * ldres + col];
                    C[(size_t)row * ldc + col] = v;
                }
            }
}

// ---------------- TDM warm: DMA leading weight tiles toward L2 via Tensor Data Mover ----------
// Builds a real D# (ISA 8.3/8.4): 2D tensor, data_size=4B, 128x32 tile, row stride = ld.
__global__ __launch_bounds__(32)
void tdm_warm_kernel(const float* __restrict__ W, int ld)
{
#if __has_builtin(__builtin_amdgcn_tensor_load_to_lds)
    __shared__ float stage[4096];   // 16 KB landing zone at LDS offset ~0
    const unsigned long long ga = (unsigned long long)(uintptr_t)W;
    const unsigned dim0 = 128, dim1 = 32;   // tile == tensor: 128 floats x 32 rows

    u32x4 g0;
    g0[0] = 1u;                                      // count=1, user mode, no gather
    g0[1] = 0u;                                      // lds_addr = 0
    g0[2] = (unsigned)(ga & 0xFFFFFFFFu);            // global_addr[31:0]  (bits 95:64)
    g0[3] = (unsigned)((ga >> 32) & 0x01FFFFFFu)     // global_addr[56:32] (bits 120:96)
          | 0x80000000u;                             // type = 2 (bits 127:126 = 10b)

    i32x8 g1;
    g1[0] = (int)(2u << 16);                         // workgroup_mask=0, data_size=2 (4B)
    g1[1] = (int)((dim0 & 0xFFFFu) << 16);           // tensor_dim0[15:0] at bits 63:48
    g1[2] = (int)(((dim0 >> 16) & 0xFFFFu)           // tensor_dim0[31:16]
          | ((dim1 & 0xFFFFu) << 16));               // tensor_dim1[15:0]
    g1[3] = (int)(((dim1 >> 16) & 0xFFFFu)
          | ((dim0 & 0xFFFFu) << 16));               // tile_dim0 = 128 (bits 127:112)
    g1[4] = (int)(dim1 & 0xFFFFu);                   // tile_dim1 = 32, tile_dim2 = 0
    g1[5] = (int)((unsigned)ld);                     // tensor_dim0_stride[31:0]
    g1[6] = 0;                                       // stride[47:32]=0, dim1_stride lo = 0
    g1[7] = 0;

    i32x4 g2; g2[0] = 0; g2[1] = 0; g2[2] = 0; g2[3] = 0;   // 2D tensor: groups 2/3 unused
    i32x4 g3; g3[0] = 0; g3[1] = 0; g3[2] = 0; g3[3] = 0;

#if defined(__clang_major__) && (__clang_major__ >= 23)
    i32x8 g4; for (int i = 0; i < 8; ++i) g4[i] = 0;
    __builtin_amdgcn_tensor_load_to_lds(g0, g1, g2, g3, g4, 0);
#else
    __builtin_amdgcn_tensor_load_to_lds(g0, g1, g2, g3, 0);
#endif
    __builtin_amdgcn_s_wait_tensorcnt(0);

    volatile float sink = stage[threadIdx.x];        // keep LDS allocated
    (void)sink;
#else
    (void)W; (void)ld;
#endif
}

// ---------------- embedding gather ----------------
__global__ __launch_bounds__(256)
void embed_kernel(const int* __restrict__ tokens, const float* __restrict__ emb,
                  float* __restrict__ x)
{
    const int idx = blockIdx.x * 256 + threadIdx.x;
    if (idx >= S_LEN * DMODEL) return;
    const int s = idx / DMODEL, d = idx - s * DMODEL;
    x[idx] = emb[(size_t)tokens[s] * DMODEL + d];
}

// ---------------- RMSNorm (one block per row) ----------------
__global__ __launch_bounds__(256)
void rmsnorm_kernel(const float* __restrict__ x, const float* __restrict__ w,
                    float* __restrict__ out, int D)
{
    __shared__ float sred[8];
    const int row = blockIdx.x;
    const int tid = threadIdx.x;
    const float* xr = x + (size_t)row * D;
    float ss = 0.0f;
    for (int i = tid; i < D; i += 256) { const float v = xr[i]; ss += v * v; }
    #pragma unroll
    for (int off = 16; off > 0; off >>= 1) ss += __shfl_down(ss, off, 32);
    if ((tid & 31) == 0) sred[tid >> 5] = ss;
    __syncthreads();
    if (tid == 0) {
        float t = 0.0f;
        #pragma unroll
        for (int i = 0; i < 8; ++i) t += sred[i];
        sred[0] = rsqrtf(t / (float)D + 1e-5f);
    }
    __syncthreads();
    const float sc = sred[0];
    float* orow = out + (size_t)row * D;
    for (int i = tid; i < D; i += 256) orow[i] = xr[i] * sc * w[i];
}

// ---------------- causal depthwise conv (width 4) + SiLU ----------------
__global__ __launch_bounds__(256)
void conv_silu_kernel(const float* __restrict__ xz, const float* __restrict__ w,
                      const float* __restrict__ b, float* __restrict__ xi)
{
    const int idx = blockIdx.x * 256 + threadIdx.x;
    if (idx >= S_LEN * DINNER) return;
    const int s = idx / DINNER, d = idx - s * DINNER;
    float acc = b[d];
    #pragma unroll
    for (int j = 0; j < DCONV; ++j) {
        const int sj = s - (DCONV - 1) + j;
        if (sj >= 0) acc += xz[(size_t)sj * (2 * DINNER) + d] * w[d * DCONV + j];
    }
    xi[idx] = silu_f(acc);
}

// ---------------- chunked selective scan ----------------
// Recurrence h[s] = a[s].*h[s-1] + dt*u*B[s],  a[s] = exp(dt*A)  (linear in h_in),
// split into NCHUNK chunks for 32x more parallelism on the sequential chain.

// Pass 1: per (chunk, channel) compute state product P = prod(a) and local end state
// h_out (with h_in = 0).
__global__ __launch_bounds__(256)
void scan_partial_kernel(const float* __restrict__ xi, const float* __restrict__ dt,
                         const float* __restrict__ proj, const float* __restrict__ A_log,
                         float* __restrict__ Pc, float* __restrict__ hout)
{
    const int d = blockIdx.x * 256 + threadIdx.x;   // channel
    const int c = blockIdx.y;                        // chunk
    if (d >= DINNER) return;

    float Arow[DSTATE], h[DSTATE], P[DSTATE];
    #pragma unroll
    for (int n = 0; n < DSTATE; ++n) {
        Arow[n] = -__expf(A_log[(size_t)d * DSTATE + n]);
        h[n] = 0.0f; P[n] = 1.0f;
    }
    const int s0 = c * SCHUNK;
    for (int s = s0; s < s0 + SCHUNK; ++s) {
        const float dtv = dt[(size_t)s * DINNER + d];
        const float du  = dtv * xi[(size_t)s * DINNER + d];
        const float* pr = proj + (size_t)s * PROJW;  // uniform across wave -> broadcast
        #pragma unroll
        for (int n = 0; n < DSTATE; ++n) {
            const float a = __expf(dtv * Arow[n]);
            h[n] = a * h[n] + du * pr[DTRANK + n];
            P[n] *= a;
        }
    }
    float* pp = Pc   + ((size_t)c * DINNER + d) * DSTATE;
    float* ph = hout + ((size_t)c * DINNER + d) * DSTATE;
    #pragma unroll
    for (int n = 0; n < DSTATE; ++n) { pp[n] = P[n]; ph[n] = h[n]; }
}

// Pass 2: tiny sequential carry across chunks per (d, n).
__global__ __launch_bounds__(256)
void scan_carry_kernel(const float* __restrict__ Pc, const float* __restrict__ hout,
                       float* __restrict__ hin)
{
    const int idx = blockIdx.x * 256 + threadIdx.x;  // (d, n) pair
    if (idx >= DINNER * DSTATE) return;
    float h = 0.0f;
    #pragma unroll 4
    for (int c = 0; c < NCHUNK; ++c) {
        const size_t o = (size_t)c * DINNER * DSTATE + idx;
        hin[o] = h;
        h = Pc[o] * h + hout[o];
    }
}

// Pass 3: re-run each chunk from its true h_in and emit y.
__global__ __launch_bounds__(256)
void scan_final_kernel(const float* __restrict__ xi, const float* __restrict__ dt,
                       const float* __restrict__ proj, const float* __restrict__ A_log,
                       const float* __restrict__ D_skip, const float* __restrict__ hin,
                       float* __restrict__ y)
{
    const int d = blockIdx.x * 256 + threadIdx.x;
    const int c = blockIdx.y;
    if (d >= DINNER) return;

    float Arow[DSTATE], h[DSTATE];
    const float* ph = hin + ((size_t)c * DINNER + d) * DSTATE;
    #pragma unroll
    for (int n = 0; n < DSTATE; ++n) {
        Arow[n] = -__expf(A_log[(size_t)d * DSTATE + n]);
        h[n] = ph[n];
    }
    const float Dv = D_skip[d];
    const int s0 = c * SCHUNK;
    for (int s = s0; s < s0 + SCHUNK; ++s) {
        const float dtv = dt[(size_t)s * DINNER + d];
        const float u   = xi[(size_t)s * DINNER + d];
        const float du  = dtv * u;
        const float* pr = proj + (size_t)s * PROJW;
        float acc = 0.0f;
        #pragma unroll
        for (int n = 0; n < DSTATE; ++n) {
            const float a = __expf(dtv * Arow[n]);
            h[n] = a * h[n] + du * pr[DTRANK + n];
            acc += h[n] * pr[DTRANK + DSTATE + n];
        }
        y[(size_t)s * DINNER + d] = acc + u * Dv;
    }
}

// ---------------- gating: g = y * silu(res) ----------------
__global__ __launch_bounds__(256)
void gate_kernel(const float* __restrict__ y, const float* __restrict__ xz,
                 float* __restrict__ g)
{
    const int idx = blockIdx.x * 256 + threadIdx.x;
    if (idx >= S_LEN * DINNER) return;
    const int s = idx / DINNER, d = idx - s * DINNER;
    g[idx] = y[idx] * silu_f(xz[(size_t)s * (2 * DINNER) + DINNER + d]);
}

// ---------------- host side ----------------
static inline int cdiv(int a, int b) { return (a + b - 1) / b; }

static void gemm_plain(const float* A, int lda, const float* B, int ldb,
                       float* C, int ldc, int M, int N, int K, hipStream_t s)
{
    dim3 g(cdiv(N, BN), cdiv(M, BM));
    gemm_tn_wmma<0, false, false><<<g, 256, 0, s>>>(A, lda, B, ldb, nullptr, nullptr, 0,
                                                    C, ldc, M, N, K);
}
static void gemm_bias_softplus(const float* A, int lda, const float* B, int ldb,
                               const float* bias, float* C, int ldc,
                               int M, int N, int K, hipStream_t s)
{
    dim3 g(cdiv(N, BN), cdiv(M, BM));
    gemm_tn_wmma<1, true, false><<<g, 256, 0, s>>>(A, lda, B, ldb, bias, nullptr, 0,
                                                   C, ldc, M, N, K);
}
static void gemm_residual(const float* A, int lda, const float* B, int ldb,
                          const float* res, int ldres, float* C, int ldc,
                          int M, int N, int K, hipStream_t s)
{
    dim3 g(cdiv(N, BN), cdiv(M, BM));
    gemm_tn_wmma<0, false, true><<<g, 256, 0, s>>>(A, lda, B, ldb, nullptr, res, ldres,
                                                   C, ldc, M, N, K);
}

extern "C" void kernel_launch(void* const* d_in, const int* in_sizes, int n_in,
                              void* d_out, int out_size, void* d_ws, size_t ws_size,
                              hipStream_t stream)
{
    (void)in_sizes; (void)n_in; (void)out_size; (void)ws_size;

    const int*   tokens      = (const int*)  d_in[0];
    const float* embedding   = (const float*)d_in[1];
    const float* norm_w      = (const float*)d_in[2];
    const float* in_proj_w   = (const float*)d_in[3];
    const float* conv_w      = (const float*)d_in[4];
    const float* conv_b      = (const float*)d_in[5];
    const float* x_proj_w    = (const float*)d_in[6];
    const float* dt_proj_w   = (const float*)d_in[7];
    const float* dt_proj_b   = (const float*)d_in[8];
    const float* A_log       = (const float*)d_in[9];
    const float* D_skip      = (const float*)d_in[10];
    const float* out_proj_w  = (const float*)d_in[11];
    const float* final_norm_w= (const float*)d_in[12];
    float*       logits      = (float*)d_out;

    // workspace carve-up (all fp32), ~54 MB total
    float* ws   = (float*)d_ws;
    float* x    = ws;  ws += (size_t)S_LEN * DMODEL;
    float* hbuf = ws;  ws += (size_t)S_LEN * DMODEL;
    float* xz   = ws;  ws += (size_t)S_LEN * 2 * DINNER;
    float* xi   = ws;  ws += (size_t)S_LEN * DINNER;
    float* proj = ws;  ws += (size_t)S_LEN * PROJW;
    float* dtb  = ws;  ws += (size_t)S_LEN * DINNER;
    float* yb   = ws;  ws += (size_t)S_LEN * DINNER;
    float* gb   = ws;  ws += (size_t)S_LEN * DINNER;
    float* Pc   = ws;  ws += (size_t)NCHUNK * DINNER * DSTATE;   // chunk state products
    float* hout = ws;  ws += (size_t)NCHUNK * DINNER * DSTATE;   // chunk local end states
    float* hin  = ws;  ws += (size_t)NCHUNK * DINNER * DSTATE;   // chunk incoming states

    // warm leading embedding tiles toward L2 via the Tensor Data Mover
    tdm_warm_kernel<<<1, 32, 0, stream>>>(embedding, DMODEL);

    embed_kernel<<<cdiv(S_LEN * DMODEL, 256), 256, 0, stream>>>(tokens, embedding, x);

    const dim3 scanGrid(cdiv(DINNER, 256), NCHUNK);

    for (int i = 0; i < NLAYERS; ++i) {
        const float* nw  = norm_w     + (size_t)i * DMODEL;
        const float* ipw = in_proj_w  + (size_t)i * 2 * DINNER * DMODEL;
        const float* cw  = conv_w     + (size_t)i * DINNER * DCONV;
        const float* cb  = conv_b     + (size_t)i * DINNER;
        const float* xpw = x_proj_w   + (size_t)i * PROJW * DINNER;
        const float* dpw = dt_proj_w  + (size_t)i * DINNER * DTRANK;
        const float* dpb = dt_proj_b  + (size_t)i * DINNER;
        const float* al  = A_log      + (size_t)i * DINNER * DSTATE;
        const float* dsk = D_skip     + (size_t)i * DINNER;
        const float* opw = out_proj_w + (size_t)i * DMODEL * DINNER;

        rmsnorm_kernel<<<S_LEN, 256, 0, stream>>>(x, nw, hbuf, DMODEL);
        gemm_plain(hbuf, DMODEL, ipw, DMODEL, xz, 2 * DINNER, S_LEN, 2 * DINNER, DMODEL, stream);
        conv_silu_kernel<<<cdiv(S_LEN * DINNER, 256), 256, 0, stream>>>(xz, cw, cb, xi);
        gemm_plain(xi, DINNER, xpw, DINNER, proj, PROJW, S_LEN, PROJW, DINNER, stream);
        gemm_bias_softplus(proj, PROJW, dpw, DTRANK, dpb, dtb, DINNER, S_LEN, DINNER, DTRANK, stream);

        // chunked scan: partial -> carry -> final
        scan_partial_kernel<<<scanGrid, 256, 0, stream>>>(xi, dtb, proj, al, Pc, hout);
        scan_carry_kernel<<<cdiv(DINNER * DSTATE, 256), 256, 0, stream>>>(Pc, hout, hin);
        scan_final_kernel<<<scanGrid, 256, 0, stream>>>(xi, dtb, proj, al, dsk, hin, yb);

        gate_kernel<<<cdiv(S_LEN * DINNER, 256), 256, 0, stream>>>(yb, xz, gb);
        gemm_residual(gb, DINNER, opw, DINNER, x, DMODEL, x, DMODEL, S_LEN, DMODEL, DINNER, stream);
    }

    rmsnorm_kernel<<<S_LEN, 256, 0, stream>>>(x, final_norm_w, hbuf, DMODEL);
    gemm_plain(hbuf, DMODEL, embedding, DMODEL, logits, NVOCAB, S_LEN, NVOCAB, DMODEL, stream);
}
